// MSA_82669530513515
// MI455X (gfx1250) — matrix operand order repaired and checked by reading, not compiled, not run on hardware
//
#include <hip/hip_runtime.h>

// Problem constants (from reference): B=4, H=W=64 -> N=4096 tokens, C=256,
// NUM_HEADS=4 -> hd=64. qkv: [C,3C]=[256,768].
#define B_   4
#define NH   4
#define NTOK 4096
#define C_   256
#define HD   64
#define C3   768

typedef __attribute__((ext_vector_type(16))) _Float16 v16h;
typedef __attribute__((ext_vector_type(8)))  _Float16 v8h;
typedef __attribute__((ext_vector_type(4)))  _Float16 v4h;
typedef __attribute__((ext_vector_type(8)))  float    v8f;
typedef __attribute__((ext_vector_type(4)))  int      v4i;
typedef __attribute__((ext_vector_type(8)))  int      v8i;
typedef __attribute__((ext_vector_type(4)))  unsigned int v4u;

#if __has_builtin(__builtin_amdgcn_global_load_async_to_lds_b128)
#define USE_ASYNC_LDS 1
#else
#define USE_ASYNC_LDS 0
#endif

#if __has_builtin(__builtin_amdgcn_tensor_load_to_lds) && \
    __has_builtin(__builtin_amdgcn_s_wait_tensorcnt)
#define USE_TDM 1
#else
#define USE_TDM 0
#endif

// Copy 16 bytes global -> LDS. Async (ASYNCcnt-tracked, bypasses VGPRs) when
// the gfx1250 builtin is available, synchronous staging otherwise.
static __device__ __forceinline__ void copy16_g2lds(_Float16* dst,
                                                    const _Float16* src) {
#if USE_ASYNC_LDS
  // Builtin signature (from hipcc diagnostic): param0 is `int4 __device__*`
  // (addrspace(1)), param1 the LDS-side pointer.  as3 pointers are 32-bit;
  // the low 32 bits of a flat LDS address are the LDS byte offset
  // (ISA 10.2), so the integer casts are value-correct.
  __builtin_amdgcn_global_load_async_to_lds_b128(
      (__attribute__((address_space(1))) v4i*)(unsigned long long)src,
      (__attribute__((address_space(3))) v4i*)(unsigned long long)dst,
      0, 0);
#else
  *(v8h*)dst = *(const v8h*)src;
#endif
}

static __device__ __forceinline__ void wait_async0() {
#if USE_ASYNC_LDS
#if __has_builtin(__builtin_amdgcn_s_wait_asynccnt)
  __builtin_amdgcn_s_wait_asynccnt(0);
#else
  asm volatile("s_wait_asynccnt 0x0" ::: "memory");
#endif
#endif
}

#if USE_TDM
// TDM 2D tile load: rows x cols f16 tile (row-major, dim0 = cols contiguous)
// from a [tensorRows][cols] f16 tensor into contiguous LDS.  Descriptor per
// CDNA5 ISA section 8 (D# groups); groups 2/3 zero => 2D tensor.
// This toolchain's builtin takes 6 args (g0, g1, g2, g3, g4pad, cpol).
static __device__ __forceinline__ void tdm_load_2d_f16(const _Float16* gsrc,
                                                       void* lds_dst, int rows,
                                                       int cols,
                                                       int tensorRows) {
  const unsigned long long ga = (unsigned long long)gsrc;
  const unsigned int lds = (unsigned int)(unsigned long long)lds_dst;
  v4u g0;
  g0[0] = 1u;                                // count=1, user mode, no gather
  g0[1] = lds;                               // lds_addr (bytes)
  g0[2] = (unsigned int)(ga & 0xffffffffu);  // global_addr[31:0]
  g0[3] = (unsigned int)((ga >> 32) & 0x1ffffffu) |
          (2u << 30);                        // global_addr[56:32] | type=2
  v8i g1;
  g1[0] = (1 << 16);                              // mask=0, data_size=1 (2B)
  g1[1] = (cols & 0xffff) << 16;                  // tensor_dim0[15:0]
  g1[2] = ((cols >> 16) & 0xffff) |               // tensor_dim0[31:16]
          ((tensorRows & 0xffff) << 16);          // tensor_dim1[15:0]
  g1[3] = ((tensorRows >> 16) & 0xffff) |         // tensor_dim1[31:16]
          ((cols & 0xffff) << 16);                // tile_dim0
  g1[4] = (rows & 0xffff);                        // tile_dim1 (tile_dim2=0)
  g1[5] = cols;                                   // tensor_dim0_stride[31:0]
  g1[6] = 0;                                      // strides hi = 0
  g1[7] = 0;
  const v4i z4 = {0, 0, 0, 0};
  const v8i z8 = {0, 0, 0, 0, 0, 0, 0, 0};
  __builtin_amdgcn_tensor_load_to_lds(g0, g1, z4, z4, z8, 0);
}
#endif

static __device__ __forceinline__ v16h cat16(v8h lo, v8h hi) {
  v16h r;
#pragma unroll
  for (int i = 0; i < 8; ++i) { r[i] = lo[i]; r[i + 8] = hi[i]; }
  return r;
}

static __device__ __forceinline__ v8f wmma_f16(v16h a, v16h b, v8f c) {
  // D = A(16x32 f16) * B(32x16 f16) + C(16x16 f32)
  return __builtin_amdgcn_wmma_f32_16x16x32_f16(
      /*neg_a=*/false, a, /*neg_b=*/false, b,
      /*c_mod=*/(short)0, c, /*reuse_a=*/false, /*reuse_b=*/false);
}

// ---------------------------------------------------------------------------
// Kernel 1: QKV projection.  qkv[row][c] = x[row][:] . wqkv[:][c] + bqkv[c]
// row = b*N + n  (M = 16384), K = 256, cols = 768.
// Scatter f16 results into q/k/v laid out [b][h][n][d]; Q pre-scaled by
// hd^-0.5 = 0.125 (scale applied after bias, matching reference q*scale).
// Block = 256 thr (8 waves), tile 64 rows x 64 cols, K step 32.
// ---------------------------------------------------------------------------
__global__ __launch_bounds__(256)
void msa_qkv_kernel(const float* __restrict__ x, const float* __restrict__ wqkv,
                    const float* __restrict__ bqkv,
                    _Float16* __restrict__ qf, _Float16* __restrict__ kf,
                    _Float16* __restrict__ vf) {
  __shared__ _Float16 At[64][32];  // [row][k]   f16 staged A
  __shared__ _Float16 Bt[64][32];  // [col][k]   f16 staged B (transposed)

  const int tid  = threadIdx.x;
  const int wave = tid >> 5, lane = tid & 31;
  const int hh   = lane >> 4, rl = lane & 15;
  const int wr   = wave >> 1, wc = wave & 1;     // 4x2 wave grid
  const int rowBase = blockIdx.x * 64;
  const int colBase = blockIdx.y * 64;

  v8f c0 = {}, c1 = {};
  for (int k0 = 0; k0 < C_; k0 += 32) {
    __syncthreads();
    {  // stage A: 64x32 floats -> f16
      const int r = tid >> 3, s = tid & 7;
#pragma unroll
      for (int p = 0; p < 2; ++p) {
        const float4 f =
            *(const float4*)&x[(size_t)(rowBase + r + 32 * p) * C_ + k0 + s * 4];
        v4h h4;
        h4[0] = (_Float16)f.x; h4[1] = (_Float16)f.y;
        h4[2] = (_Float16)f.z; h4[3] = (_Float16)f.w;
        *(v4h*)&At[r + 32 * p][s * 4] = h4;
      }
    }
    {  // stage B transposed: Bt[c][k] = wqkv[k0+k][colBase+c]
      const int kk = tid >> 4, s2 = tid & 15;
#pragma unroll
      for (int p = 0; p < 2; ++p) {
        const float4 f = *(const float4*)&wqkv[(size_t)(k0 + kk + 16 * p) * C3 +
                                               colBase + s2 * 4];
        Bt[s2 * 4 + 0][kk + 16 * p] = (_Float16)f.x;
        Bt[s2 * 4 + 1][kk + 16 * p] = (_Float16)f.y;
        Bt[s2 * 4 + 2][kk + 16 * p] = (_Float16)f.z;
        Bt[s2 * 4 + 3][kk + 16 * p] = (_Float16)f.w;
      }
    }
    __syncthreads();

    // A fragment: lane holds row (lane&15), K = 16*(lane>=16)+h contiguous
    const v16h A = cat16(*(const v8h*)&At[wr * 16 + rl][hh * 16],
                         *(const v8h*)&At[wr * 16 + rl][hh * 16 + 8]);
    // B fragments: lane holds column, K = 16*(lane>=16)+h contiguous
    const v16h Bf0 = cat16(*(const v8h*)&Bt[wc * 32 + rl][hh * 16],
                           *(const v8h*)&Bt[wc * 32 + rl][hh * 16 + 8]);
    const v16h Bf1 = cat16(*(const v8h*)&Bt[wc * 32 + 16 + rl][hh * 16],
                           *(const v8h*)&Bt[wc * 32 + 16 + rl][hh * 16 + 8]);
    c0 = wmma_f16(A, Bf0, c0);
    c1 = wmma_f16(A, Bf1, c1);
  }

  // Epilogue: bias + scatter into q/k/v f16 [b][h][n][d]
#pragma unroll
  for (int t = 0; t < 2; ++t) {
    const v8f cc  = t ? c1 : c0;
    const int col = colBase + wc * 32 + t * 16 + rl;
    const int which = col >> 8;       // 0=q, 1=k, 2=v
    const int rem   = col & 255;
    const int head  = rem >> 6;
    const int d     = rem & 63;
    const float bias = bqkv[col];
    _Float16* dst   = (which == 0) ? qf : ((which == 1) ? kf : vf);
    const float mul = (which == 0) ? 0.125f : 1.0f;  // hd^-0.5 folded into Q
#pragma unroll
    for (int i = 0; i < 8; ++i) {
      const int row = rowBase + wr * 16 + i + 8 * hh;  // C/D frag: M=i+8*hh
      const int b = row >> 12, n = row & (NTOK - 1);
      const float val = (cc[i] + bias) * mul;
      dst[(((size_t)(b * NH + head)) * NTOK + n) * HD + d] = (_Float16)val;
    }
  }
}

// ---------------------------------------------------------------------------
// Kernel 2: flash attention per (b,h).  Each block: 128 q rows (16 per wave).
// Per 32-key step: K tile DMA'd to LDS by the Tensor Data Mover (one
// tensor_load_to_lds from wave 0, TENSORcnt-tracked; async-copy fallback),
// V tile staged transposed, S = Q.K^T via 4 wmma, online softmax with
// half-wave shfl reductions, P transposed through per-wave LDS into an
// A-fragment, then O += P.V via 4 wmma.  Next step's K/V rows are prefetched
// (global_prefetch_b8) while this step computes.
// ---------------------------------------------------------------------------
__global__ __launch_bounds__(256)
void msa_attn_kernel(const _Float16* __restrict__ qf,
                     const _Float16* __restrict__ kf,
                     const _Float16* __restrict__ vf,
                     _Float16* __restrict__ of) {
  __shared__ _Float16 Ks[32][64];     // [j][d]  K tile, natural layout
  __shared__ _Float16 Vt[64][32];     // [d][j]  V tile transposed
  __shared__ _Float16 Pb[8][16][32];  // per-wave P tile (D-frag -> A-frag)

  const int tid  = threadIdx.x;
  const int wave = tid >> 5, lane = tid & 31;
  const int hh   = lane >> 4, rl = lane & 15;
  const int bh    = blockIdx.y;
  const int qBase = blockIdx.x * 128;

  const _Float16* Qp = qf + (size_t)bh * NTOK * HD;
  const _Float16* Kp = kf + (size_t)bh * NTOK * HD;
  const _Float16* Vp = vf + (size_t)bh * NTOK * HD;

  // Q fragments, resident for the whole kernel (hd=64 -> 2 K-chunks of 32)
  const int qrow = qBase + wave * 16 + rl;
  v16h Qf2[2];
#pragma unroll
  for (int ch = 0; ch < 2; ++ch)
    Qf2[ch] = cat16(*(const v8h*)&Qp[(size_t)qrow * HD + ch * 32 + hh * 16],
                    *(const v8h*)&Qp[(size_t)qrow * HD + ch * 32 + hh * 16 + 8]);

  float m[8], lsum[8];
  v8f acc[4];
  const v8f vzero = {};
#pragma unroll
  for (int i = 0; i < 8; ++i) { m[i] = -3.0e38f; lsum[i] = 0.0f; }
#pragma unroll
  for (int t = 0; t < 4; ++t) acc[t] = vzero;

  for (int j0 = 0; j0 < NTOK; j0 += 32) {
    __syncthreads();
#if USE_TDM
    // One TDM descriptor moves the whole 32x64 f16 K tile into LDS.
    if (wave == 0)
      tdm_load_2d_f16(&Kp[(size_t)j0 * HD], &Ks[0][0], 32, HD, NTOK);
#endif
    {
      const int jj = tid >> 3, s = tid & 7;
#if !USE_TDM
      // K tile -> LDS (async copy, natural [j][d] layout; one b128 per thread)
      copy16_g2lds(&Ks[jj][s * 8], &Kp[(size_t)(j0 + jj) * HD + s * 8]);
#endif
      // V tile -> LDS transposed: Vt[d][jj] = V[j0+jj][d]
      const v8h vv = *(const v8h*)&Vp[(size_t)(j0 + jj) * HD + s * 8];
#pragma unroll
      for (int e = 0; e < 8; ++e) Vt[s * 8 + e][jj] = vv[e];
      // Prefetch next step's rows into cache while we compute this one.
      const int jn = (j0 + 32 < NTOK) ? (j0 + 32) : j0;
      __builtin_prefetch(&Kp[(size_t)(jn + jj) * HD + s * 8], 0, 2);
      __builtin_prefetch(&Vp[(size_t)(jn + jj) * HD + s * 8], 0, 2);
    }
#if USE_TDM
    if (wave == 0) __builtin_amdgcn_s_wait_tensorcnt(0);
#else
    wait_async0();
#endif
    __syncthreads();

    // ---- S = Q.K^T : two 16x16 j-subtiles, K-dim = hd (2 chunks of 32)
    v8f S0 = vzero, S1 = vzero;
#pragma unroll
    for (int ch = 0; ch < 2; ++ch) {
      {
        const v16h Bf = cat16(*(const v8h*)&Ks[rl][ch * 32 + hh * 16],
                              *(const v8h*)&Ks[rl][ch * 32 + hh * 16 + 8]);
        S0 = wmma_f16(Qf2[ch], Bf, S0);
      }
      {
        const v16h Bf = cat16(*(const v8h*)&Ks[16 + rl][ch * 32 + hh * 16],
                              *(const v8h*)&Ks[16 + rl][ch * 32 + hh * 16 + 8]);
        S1 = wmma_f16(Qf2[ch], Bf, S1);
      }
    }

    // ---- online softmax (row M = i + 8*hh lives in one 16-lane half)
#pragma unroll
    for (int i = 0; i < 8; ++i) {
      float t = fmaxf(S0[i], S1[i]);
      t = fmaxf(t, __shfl_xor(t, 1));
      t = fmaxf(t, __shfl_xor(t, 2));
      t = fmaxf(t, __shfl_xor(t, 4));
      t = fmaxf(t, __shfl_xor(t, 8));
      const float mn = fmaxf(m[i], t);
      const float sc = __expf(m[i] - mn);
      const float p0 = __expf(S0[i] - mn);
      const float p1 = __expf(S1[i] - mn);
      float s = p0 + p1;
      s += __shfl_xor(s, 1);
      s += __shfl_xor(s, 2);
      s += __shfl_xor(s, 4);
      s += __shfl_xor(s, 8);
      lsum[i] = lsum[i] * sc + s;
      m[i] = mn;
#pragma unroll
      for (int t4 = 0; t4 < 4; ++t4) acc[t4][i] *= sc;
      // D-frag element (row i+8*hh, col rl / rl+16) -> LDS for transpose
      Pb[wave][i + 8 * hh][rl]      = (_Float16)p0;
      Pb[wave][i + 8 * hh][rl + 16] = (_Float16)p1;
    }
    // Per-wave LDS ops are in-order; fence the compiler and drain DScnt.
    asm volatile("s_wait_dscnt 0x0" ::: "memory");
    // Reload P as an A-fragment (row rl, K = 16*hh + 0..15 contiguous)
    const v16h Pf = cat16(*(const v8h*)&Pb[wave][rl][hh * 16],
                          *(const v8h*)&Pb[wave][rl][hh * 16 + 8]);

    // ---- O += P.V  (4 d-subtiles of 16)
#pragma unroll
    for (int t4 = 0; t4 < 4; ++t4) {
      const v16h Bf = cat16(*(const v8h*)&Vt[t4 * 16 + rl][hh * 16],
                            *(const v8h*)&Vt[t4 * 16 + rl][hh * 16 + 8]);
      acc[t4] = wmma_f16(Pf, Bf, acc[t4]);
    }
  }

  // Epilogue: normalize and merge heads into o f16 [b][n][C]
  const int h = bh & (NH - 1), b = bh >> 2;
#pragma unroll
  for (int t4 = 0; t4 < 4; ++t4) {
    const int d = t4 * 16 + rl;
#pragma unroll
    for (int i = 0; i < 8; ++i) {
      const int n = qBase + wave * 16 + i + 8 * hh;
      const float val = acc[t4][i] / lsum[i];
      of[((size_t)(b * NTOK + n)) * C_ + h * HD + d] = (_Float16)val;
    }
  }
}

// ---------------------------------------------------------------------------
// Kernel 3: out projection.  out[row][c] = o[row][:] . wout[:][c] + bout[c]
// A is already f16 (attention output) -> async-copied straight into LDS.
// ---------------------------------------------------------------------------
__global__ __launch_bounds__(256)
void msa_outproj_kernel(const _Float16* __restrict__ of,
                        const float* __restrict__ wout,
                        const float* __restrict__ bout,
                        float* __restrict__ out) {
  __shared__ _Float16 At[64][32];
  __shared__ _Float16 Bt[64][32];

  const int tid  = threadIdx.x;
  const int wave = tid >> 5, lane = tid & 31;
  const int hh   = lane >> 4, rl = lane & 15;
  const int wr   = wave >> 1, wc = wave & 1;
  const int rowBase = blockIdx.x * 64;
  const int colBase = blockIdx.y * 64;

  v8f c0 = {}, c1 = {};
  for (int k0 = 0; k0 < C_; k0 += 32) {
    __syncthreads();
    {  // stage A (f16, straight copy): one async b128 per thread
      const int r = tid >> 2, s = tid & 3;
      copy16_g2lds(&At[r][s * 8], &of[(size_t)(rowBase + r) * C_ + k0 + s * 8]);
    }
    {  // stage B transposed from f32 wout
      const int kk = tid >> 4, s2 = tid & 15;
#pragma unroll
      for (int p = 0; p < 2; ++p) {
        const float4 f = *(const float4*)&wout[(size_t)(k0 + kk + 16 * p) * C_ +
                                               colBase + s2 * 4];
        Bt[s2 * 4 + 0][kk + 16 * p] = (_Float16)f.x;
        Bt[s2 * 4 + 1][kk + 16 * p] = (_Float16)f.y;
        Bt[s2 * 4 + 2][kk + 16 * p] = (_Float16)f.z;
        Bt[s2 * 4 + 3][kk + 16 * p] = (_Float16)f.w;
      }
    }
    wait_async0();
    __syncthreads();

    const v16h A = cat16(*(const v8h*)&At[wr * 16 + rl][hh * 16],
                         *(const v8h*)&At[wr * 16 + rl][hh * 16 + 8]);
    const v16h Bf0 = cat16(*(const v8h*)&Bt[wc * 32 + rl][hh * 16],
                           *(const v8h*)&Bt[wc * 32 + rl][hh * 16 + 8]);
    const v16h Bf1 = cat16(*(const v8h*)&Bt[wc * 32 + 16 + rl][hh * 16],
                           *(const v8h*)&Bt[wc * 32 + 16 + rl][hh * 16 + 8]);
    c0 = wmma_f16(A, Bf0, c0);
    c1 = wmma_f16(A, Bf1, c1);
  }

#pragma unroll
  for (int t = 0; t < 2; ++t) {
    const v8f cc  = t ? c1 : c0;
    const int col = colBase + wc * 32 + t * 16 + rl;
    const float bias = bout[col];
#pragma unroll
    for (int i = 0; i < 8; ++i) {
      const int row = rowBase + wr * 16 + i + 8 * hh;
      out[(size_t)row * C_ + col] = cc[i] + bias;
    }
  }
}

// ---------------------------------------------------------------------------
extern "C" void kernel_launch(void* const* d_in, const int* in_sizes, int n_in,
                              void* d_out, int out_size, void* d_ws,
                              size_t ws_size, hipStream_t stream) {
  const float* x    = (const float*)d_in[0];  // [4,64,64,256]
  const float* wqkv = (const float*)d_in[1];  // [256,768]
  const float* bqkv = (const float*)d_in[2];  // [768]
  const float* wout = (const float*)d_in[3];  // [256,256]
  const float* bout = (const float*)d_in[4];  // [256]
  float* out = (float*)d_out;                 // [4,64,64,256] f32

  // Workspace: q,k,v f16 [b][h][n][d] (8 MB each) + o f16 [b][n][C] (8 MB)
  const size_t nPerTensor = (size_t)B_ * NH * NTOK * HD;  // 4,194,304 halves
  _Float16* qf = (_Float16*)d_ws;
  _Float16* kf = qf + nPerTensor;
  _Float16* vf = kf + nPerTensor;
  _Float16* of = vf + nPerTensor;

  msa_qkv_kernel<<<dim3(B_ * NTOK / 64, C3 / 64), 256, 0, stream>>>(
      x, wqkv, bqkv, qf, kf, vf);
  msa_attn_kernel<<<dim3(NTOK / 128, B_ * NH), 256, 0, stream>>>(
      qf, kf, vf, of);
  msa_outproj_kernel<<<dim3(B_ * NTOK / 64, C_ / 64), 256, 0, stream>>>(
      of, wout, bout, out);
}
